// AttentionTorch_77833397338547
// MI455X (gfx1250) — matrix operand order repaired and checked by reading, not compiled, not run on hardware
//
#include <hip/hip_runtime.h>

// ---------------------------------------------------------------------------
// MI455X (gfx1250) attention + output projection.
// Roofline: ~69 GFLOP total vs ~150MB of HBM traffic (~6.5us @ 23.3TB/s).
// bf16 WMMA (f32 accumulate) keeps the matmuls near-balanced with HBM; all
// per-(b,h) K/V slices, W_out, and the bf16 hid tensor are L2-resident
// (192MB), so cross-tile re-reads hit L2, not HBM.
// hid is stored TRANSPOSED as [b][q][c] so (a) the attention epilogue packs
// 8 consecutive-d accumulators into 16B stores, and (b) the projection's
// B fragments ([n=q][K=c], K-contiguous) load straight from global memory
// with b128 loads -- no LDS re-transpose stage at all.
// ---------------------------------------------------------------------------

typedef __attribute__((ext_vector_type(16))) __bf16 v16bf;
typedef __attribute__((ext_vector_type(8)))  __bf16 v8bf;
typedef __attribute__((ext_vector_type(8)))  float  v8f;
typedef unsigned short u16;
typedef unsigned int   u32;

#define NB 4
#define NC 2048
#define NS 1024
#define NH 16
#define ND 128
#define SCALE 0.08838834764831845f   // 1/sqrt(128)

// fp32 -> bf16 round-to-nearest-even
__device__ __forceinline__ u16 f2bf(float x) {
  u32 u = __float_as_uint(x);
  return (u16)((u + 0x7FFFu + ((u >> 16) & 1u)) >> 16);
}
__device__ __forceinline__ u32 pack2(float a, float b) {
  return (u32)f2bf(a) | ((u32)f2bf(b) << 16);
}

// A-fragment (16x32 bf16): lane holds row m, K = off+{0..7} and off+16+{0..7}
__device__ __forceinline__ v16bf ldA(const u16* row, int off) {
  v8bf lo = *(const v8bf*)(row + off);
  v8bf hi = *(const v8bf*)(row + off + 16);
  return __builtin_shufflevector(lo, hi, 0,1,2,3,4,5,6,7,8,9,10,11,12,13,14,15);
}
// B-fragment (32x16 bf16): lane holds col n, K = off+{0..7} and off+16+{0..7}
__device__ __forceinline__ v16bf ldB(const u16* p) {
  v8bf lo = *(const v8bf*)(p);
  v8bf hi = *(const v8bf*)(p + 16);
  return __builtin_shufflevector(lo, hi, 0,1,2,3,4,5,6,7,8,9,10,11,12,13,14,15);
}
__device__ __forceinline__ v8f wmma_bf16(v16bf a, v16bf b, v8f c) {
  return __builtin_amdgcn_wmma_f32_16x16x32_bf16(false, a, false, b, (short)0, c,
                                                 false, false);
}
#define LDS_FENCE() asm volatile("s_wait_dscnt 0" ::: "memory")

// LDS row strides (bf16 elements): multiples of 8 for 16B-aligned
// ds_load_b128 fragment reads, padded to break power-of-two bank patterns.
#define QT_STRIDE 144
#define KT_STRIDE 136
#define VT_STRIDE 40
#define PT_STRIDE 48

// ---------------------------------------------------------------------------
// Kernel 1: flash-style attention, one wave per (b, h, 16-wide q tile).
// hid (bf16) written to workspace TRANSPOSED: [b][q][c = h*128+d].
// ---------------------------------------------------------------------------
__global__ __launch_bounds__(32)
void attn_kernel(const float* __restrict__ keys,
                 const float* __restrict__ values,
                 const float* __restrict__ queries,
                 const int* __restrict__ mask,
                 u16* __restrict__ hid) {
  __shared__ u16 Qt[16 * QT_STRIDE];   // [q][d]   B-layout for scores
  __shared__ u16 Kt[32 * KT_STRIDE];   // [k][d]   A-layout for scores (transposed stage)
  __shared__ u16 Vt[ND * VT_STRIDE];   // [d][k]   A-layout for P*V (direct stage)
  __shared__ u16 Pt[16 * PT_STRIDE];   // [q][k]   B-layout for P*V

  const int q0   = blockIdx.x * 16;
  const int h    = blockIdx.y;
  const int b    = blockIdx.z;
  const int lane = threadIdx.x;
  const int lq   = lane & 15;          // C-fragment column / fragment row
  const int hf   = lane >> 4;          // half-wave select

  const float* Qg = queries + ((size_t)b * NC + (size_t)h * ND) * NS; // [d][s]
  const float* Kg = keys    + ((size_t)b * NC + (size_t)h * ND) * NS;
  const float* Vg = values  + ((size_t)b * NC + (size_t)h * ND) * NS;
  const int*   Mg = mask + (size_t)b * NS;

  // ---- stage Q tile (transpose [d][q] -> [q][d], convert to bf16) ----
  {
    const int dbase = hf * 64;   // lanes 0-15: d 0..63 for row q=lq; 16-31: d 64..127
#pragma unroll 8
    for (int dd = 0; dd < 64; dd += 2) {
      float x0 = Qg[(size_t)(dbase + dd)     * NS + q0 + lq];
      float x1 = Qg[(size_t)(dbase + dd + 1) * NS + q0 + lq];
      *(u32*)&Qt[lq * QT_STRIDE + dbase + dd] = pack2(x0, x1);
    }
  }
  LDS_FENCE();

  v8f acc[8];
#pragma unroll
  for (int t = 0; t < 8; ++t) acc[t] = (v8f){0.f,0.f,0.f,0.f,0.f,0.f,0.f,0.f};
  float m_run = -1e30f, l_run = 0.f;

  for (int k0 = 0; k0 < NS; k0 += 32) {
    // ---- prefetch next k tile into the cache hierarchy (global_prefetch_b8)
    if (k0 + 32 < NS) {
#pragma unroll
      for (int rr = 0; rr < 4; ++rr) {
        __builtin_prefetch(Kg + (size_t)(rr * 32 + lane) * NS + k0 + 32, 0, 0);
        __builtin_prefetch(Vg + (size_t)(rr * 32 + lane) * NS + k0 + 32, 0, 0);
      }
    }
    // ---- stage K tile: transpose [d][k] -> Kt[k][d], k = lane ----
#pragma unroll 8
    for (int d = 0; d < ND; d += 2) {
      float x0 = Kg[(size_t)d       * NS + k0 + lane];
      float x1 = Kg[(size_t)(d + 1) * NS + k0 + lane];
      *(u32*)&Kt[lane * KT_STRIDE + d] = pack2(x0, x1);
    }
    // ---- stage V tile: direct [d][k] -> Vt[d][k], 4 rows per lane ----
#pragma unroll
    for (int rr = 0; rr < 4; ++rr) {
      const int d = rr * 32 + lane;
      const float4* src = (const float4*)(Vg + (size_t)d * NS + k0);
#pragma unroll
      for (int kk = 0; kk < 32; kk += 4) {
        float4 x = src[kk >> 2];
        *(u32*)&Vt[d * VT_STRIDE + kk]     = pack2(x.x, x.y);
        *(u32*)&Vt[d * VT_STRIDE + kk + 2] = pack2(x.z, x.w);
      }
    }
    LDS_FENCE();

    // ---- scores: two 16(k)x16(q) tiles, contraction over d=128 ----
    v8f c0 = (v8f){0.f,0.f,0.f,0.f,0.f,0.f,0.f,0.f};
    v8f c1 = (v8f){0.f,0.f,0.f,0.f,0.f,0.f,0.f,0.f};
#pragma unroll
    for (int dblk = 0; dblk < 4; ++dblk) {
      v16bf bq = ldB(&Qt[lq * QT_STRIDE + dblk * 32 + 8 * hf]);
      v16bf a0 = ldA(&Kt[lq        * KT_STRIDE], dblk * 32 + 8 * hf);
      v16bf a1 = ldA(&Kt[(16 + lq) * KT_STRIDE], dblk * 32 + 8 * hf);
      c0 = wmma_bf16(a0, bq, c0);
      c1 = wmma_bf16(a1, bq, c1);
    }

    // ---- mask + online softmax (over k), cross-half reduce via shfl_xor 16 ----
    float s0[8], s1[8];
    bool  mk0[8], mk1[8];
    float tmax = -1e30f;
#pragma unroll
    for (int r = 0; r < 8; ++r) {
      const int kk = k0 + r + 8 * hf;      // tile0 global k; tile1 = kk+16
      mk0[r] = (Mg[kk]      != 0);
      mk1[r] = (Mg[kk + 16] != 0);
      s0[r] = c0[r] * SCALE;
      s1[r] = c1[r] * SCALE;
      tmax = fmaxf(tmax, fmaxf(mk0[r] ? s0[r] : -1e30f,
                               mk1[r] ? s1[r] : -1e30f));
    }
    tmax = fmaxf(tmax, __shfl_xor(tmax, 16, 32));
    const float m_new = fmaxf(m_run, tmax);
    const float alpha = __expf(m_run - m_new);
    l_run *= alpha;
#pragma unroll
    for (int t = 0; t < 8; ++t)
#pragma unroll
      for (int r = 0; r < 8; ++r) acc[t][r] *= alpha;

    float psum = 0.f;
#pragma unroll
    for (int r = 0; r < 8; ++r) {
      float p0 = mk0[r] ? __expf(s0[r] - m_new) : 0.f;
      float p1 = mk1[r] ? __expf(s1[r] - m_new) : 0.f;
      psum += p0 + p1;
      Pt[lq * PT_STRIDE + (r + 8 * hf)]      = f2bf(p0);  // klocal = r+8hf
      Pt[lq * PT_STRIDE + (16 + r + 8 * hf)] = f2bf(p1);  // klocal+16
    }
    psum += __shfl_xor(psum, 16, 32);
    l_run += psum;
    m_run = m_new;
    LDS_FENCE();

    // ---- P*V: 8 d-tiles of 16, contraction over klocal=32 ----
    v16bf bp = ldB(&Pt[lq * PT_STRIDE + 8 * hf]);
#pragma unroll
    for (int t = 0; t < 8; ++t) {
      v16bf av = ldA(&Vt[(t * 16 + lq) * VT_STRIDE], 8 * hf);
      acc[t] = wmma_bf16(av, bp, acc[t]);
    }
    LDS_FENCE();   // fragments consumed before next iteration restages LDS
  }

  // ---- epilogue: normalize, pack 8 consecutive-d values, 16B stores ----
  // hid layout [b][q][c]: lane owns row q = q0+lq; per tile t the 8 accum
  // elements are d = t*16 + 8*hf + {0..7}, i.e. contiguous c.
  const float inv_l = 1.0f / l_run;
  u16* hrow = hid + ((size_t)b * NS + q0 + lq) * NC + (size_t)h * ND;
#pragma unroll
  for (int t = 0; t < 8; ++t) {
    uint4 pk;
    pk.x = pack2(acc[t][0] * inv_l, acc[t][1] * inv_l);
    pk.y = pack2(acc[t][2] * inv_l, acc[t][3] * inv_l);
    pk.z = pack2(acc[t][4] * inv_l, acc[t][5] * inv_l);
    pk.w = pack2(acc[t][6] * inv_l, acc[t][7] * inv_l);
    *(uint4*)&hrow[t * 16 + 8 * hf] = pk;
  }
}

// ---------------------------------------------------------------------------
// Kernel 2: out[b,o,q] = sum_c W[o,c] * hid[b,q,c].
// One wave per 64(o) x 16(q) tile; 4 WMMAs per 32-deep c step.
// B fragments load directly from global (hid is [q][c], K-contiguous).
// ---------------------------------------------------------------------------
__global__ __launch_bounds__(32)
void proj_kernel(const float* __restrict__ wout,
                 const u16* __restrict__ hid,
                 float* __restrict__ out) {
  __shared__ u16 Wt[64 * VT_STRIDE];   // [o][c]  A-layout (direct stage)

  const int q0   = blockIdx.x * 16;
  const int o0   = blockIdx.y * 64;
  const int b    = blockIdx.z;
  const int lane = threadIdx.x;
  const int lq   = lane & 15;
  const int hf   = lane >> 4;

  // this lane's B-fragment source row: q = q0+lq, K base = 8*hf
  const u16* hrow = hid + ((size_t)b * NS + q0 + lq) * NC + 8 * hf;

  v8f acc[4];
#pragma unroll
  for (int t = 0; t < 4; ++t) acc[t] = (v8f){0.f,0.f,0.f,0.f,0.f,0.f,0.f,0.f};

  for (int c0 = 0; c0 < NC; c0 += 32) {
    // B fragment straight from global memory: two 16B chunks per lane, 32B apart
    v16bf bh = ldB(hrow + c0);

    // stage W tile: 64 o-rows x 32 c, c contiguous in memory -> direct
#pragma unroll
    for (int rr = 0; rr < 2; ++rr) {
      const int o = rr * 32 + lane;
      const float4* src = (const float4*)(wout + (size_t)(o0 + o) * NC + c0);
#pragma unroll
      for (int kk = 0; kk < 32; kk += 4) {
        float4 x = src[kk >> 2];
        *(u32*)&Wt[o * VT_STRIDE + kk]     = pack2(x.x, x.y);
        *(u32*)&Wt[o * VT_STRIDE + kk + 2] = pack2(x.z, x.w);
      }
    }
    LDS_FENCE();

#pragma unroll
    for (int mt = 0; mt < 4; ++mt) {
      v16bf aw = ldA(&Wt[(mt * 16 + lq) * VT_STRIDE], 8 * hf);
      acc[mt] = wmma_bf16(aw, bh, acc[mt]);
    }
    LDS_FENCE();   // A fragments consumed before next restage
  }

#pragma unroll
  for (int mt = 0; mt < 4; ++mt) {
#pragma unroll
    for (int r = 0; r < 8; ++r) {
      const int o = o0 + mt * 16 + r + 8 * hf;
      out[((size_t)b * NC + o) * NS + q0 + lq] = acc[mt][r];
    }
  }
}

// ---------------------------------------------------------------------------
extern "C" void kernel_launch(void* const* d_in, const int* in_sizes, int n_in,
                              void* d_out, int out_size, void* d_ws, size_t ws_size,
                              hipStream_t stream) {
  const float* keys    = (const float*)d_in[0];
  const float* values  = (const float*)d_in[1];
  const float* queries = (const float*)d_in[2];
  const int*   mask    = (const int*)d_in[3];
  const float* wout    = (const float*)d_in[4];
  float*       out     = (float*)d_out;
  u16*         hid     = (u16*)d_ws;   // B*S*C bf16 = 16 MB scratch, [b][q][c]

  dim3 g1(NS / 16, NH, NB);            // 4096 waves: (b, h, q-tile)
  attn_kernel<<<g1, 32, 0, stream>>>(keys, values, queries, mask, hid);

  dim3 g2(NS / 16, NC / 64, NB);       // 8192 waves: (b, o-tile, q-tile)
  proj_kernel<<<g2, 32, 0, stream>>>(wout, hid, out);
}
